// TttBaseModule_2370821947411
// MI455X (gfx1250) — matrix operand order repaired and checked by reading, not compiled, not run on hardware
//
#include <hip/hip_runtime.h>
#include <math.h>

// Problem constants (match reference)
#define B_ 4
#define L_ 2048
#define C_ 2048
#define NH 32
#define HD 64
#define CK 16
#define NCHUNK (L_ / CK)
#define EPSV 1e-6f

typedef __attribute__((ext_vector_type(2)))  float  v2f;
typedef __attribute__((ext_vector_type(8)))  float  v8f;
typedef __attribute__((ext_vector_type(4)))  __bf16 v4bf;
typedef __attribute__((ext_vector_type(8)))  __bf16 v8bf;
typedef __attribute__((ext_vector_type(16))) __bf16 v16bf;

// ---- CDNA5 WMMA wrappers ---------------------------------------------------
__device__ __forceinline__ v8f wmma4(v2f a, v2f b, v8f c) {
  return __builtin_amdgcn_wmma_f32_16x16x4_f32(false, a, false, b, (short)0, c,
                                               false, false);
}
__device__ __forceinline__ v8f wmma_bf16(v16bf a, v16bf b, v8f c) {
  return __builtin_amdgcn_wmma_f32_16x16x32_bf16(false, a, false, b, (short)0,
                                                 c, false, false);
}
__device__ __forceinline__ v16bf cat8(v8bf a, v8bf b) {
  return __builtin_shufflevector(a, b, 0, 1, 2, 3, 4, 5, 6, 7, 8, 9, 10, 11,
                                 12, 13, 14, 15);
}
__device__ __forceinline__ float wave_sum32(float v) {
#pragma unroll
  for (int off = 16; off > 0; off >>= 1) v += __shfl_xor(v, off, 32);
  return v;
}

// ---------------------------------------------------------------------------
// Split fp32 -> (bf16 hi, bf16 lo) planes.  x = hi + lo to ~2^-17 relative.
// ---------------------------------------------------------------------------
__global__ __launch_bounds__(256) void split_bf16(const float* __restrict__ in,
                                                  __bf16* __restrict__ hi,
                                                  __bf16* __restrict__ lo,
                                                  int n) {
  const int i = (blockIdx.x * 256 + threadIdx.x) * 4;
  if (i >= n) return;
  const float4 v = *(const float4*)(in + i);
  v4bf h, l;
  h[0] = (__bf16)v.x; l[0] = (__bf16)(v.x - (float)h[0]);
  h[1] = (__bf16)v.y; l[1] = (__bf16)(v.y - (float)h[1]);
  h[2] = (__bf16)v.z; l[2] = (__bf16)(v.z - (float)h[2]);
  h[3] = (__bf16)v.w; l[3] = (__bf16)(v.w - (float)h[3]);
  *(v4bf*)(hi + i) = h;
  *(v4bf*)(lo + i) = l;
}

// ---------------------------------------------------------------------------
// One-time weight transpose + split: W[K][N] f32 -> Wt hi/lo [N][K] bf16.
// Moves the GEMM's B-transpose out of the hot loop (weights are reused by
// all 128 M-blocks, so the GEMM can then stage both operands with pure
// b128 LDS traffic). Tiled 32x32 through LDS, both sides coalesced.
// ---------------------------------------------------------------------------
__global__ __launch_bounds__(256) void transpose_split_bf16(
    const float* __restrict__ W, __bf16* __restrict__ Th,
    __bf16* __restrict__ Tl, int K, int N) {
  __shared__ float tile[32][33];
  const int tx = threadIdx.x & 31;
  const int ty = threadIdx.x >> 5;  // 0..7
  const int k0 = blockIdx.y * 32;
  const int n0 = blockIdx.x * 32;
#pragma unroll
  for (int i = 0; i < 4; ++i) {
    const int r = ty + i * 8;
    tile[r][tx] = W[(size_t)(k0 + r) * N + n0 + tx];
  }
  __syncthreads();
#pragma unroll
  for (int i = 0; i < 4; ++i) {
    const int r = ty + i * 8;
    const float x = tile[tx][r];  // = W[k0+tx][n0+r]
    const __bf16 h = (__bf16)x;
    Th[(size_t)(n0 + r) * K + k0 + tx] = h;
    Tl[(size_t)(n0 + r) * K + k0 + tx] = (__bf16)(x - (float)h);
  }
}

// ---------------------------------------------------------------------------
// bf16x3 WMMA GEMM: C[M,N] (f32) = (Ah+Al)[M,K] @ (Bth+Btl)^T, where Bt is the
// PRE-TRANSPOSED weight [N][K].  C ~= Ah*Bh + Ah*Bl + Al*Bh in fp32
// accumulators (3 WMMAs per K=32 vs 8 fp32 WMMAs -> ~2.7x fewer matrix ops).
// Double-buffered LDS: prefetch tile i+1 to regs, 12 WMMAs, store to the
// other buffer, single barrier per K-iteration. All LDS traffic is b128.
// Block tile 64x128, 256 threads = 8 waves (2x4), each wave 32x32 (2x2 tiles).
// ---------------------------------------------------------------------------
#define BM 64
#define BN 128
#define KB2 32
#define ASTR (KB2 + 8)  // 40 bf16 = 80B row stride: 16B aligned, 20-bank step

__global__ __launch_bounds__(256) void gemm_wmma_bf16x3(
    const __bf16* __restrict__ Ah, const __bf16* __restrict__ Al,
    const __bf16* __restrict__ Bth, const __bf16* __restrict__ Btl,
    float* __restrict__ Cm, int M, int N, int K) {
  __shared__ __bf16 Ash[2][BM][ASTR];
  __shared__ __bf16 Asl[2][BM][ASTR];
  __shared__ __bf16 Bsh[2][BN][ASTR];  // [n][k] (weight already transposed)
  __shared__ __bf16 Bsl[2][BN][ASTR];

  const int tid  = threadIdx.x;
  const int wave = tid >> 5;
  const int lane = tid & 31;
  const int wm = wave >> 2;   // 0..1
  const int wn = wave & 3;    // 0..3
  const int h8 = lane >> 4;   // K-half select
  const int nl = lane & 15;   // M (A) / N (B) within 16x16 tile
  const int m0 = blockIdx.y * BM;
  const int n0 = blockIdx.x * BN;

  // staging assignments (all 16B chunks)
  const int ar = tid >> 2, ac = (tid & 3) * 8;   // A: 64 rows x 32
  const int br = tid >> 1, bc = (tid & 1) * 16;  // B: 128 rows x 32

  const v8f vzero = {0.f, 0.f, 0.f, 0.f, 0.f, 0.f, 0.f, 0.f};
  v8f acc[2][2];
#pragma unroll
  for (int mi = 0; mi < 2; ++mi)
#pragma unroll
    for (int ni = 0; ni < 2; ++ni) acc[mi][ni] = vzero;

  v8bf ah_r, al_r, bh_r0, bh_r1, bl_r0, bl_r1;
  auto load_tile = [&](int k0) {
    const size_t ga = (size_t)(m0 + ar) * K + k0 + ac;
    ah_r = *(const v8bf*)(Ah + ga);
    al_r = *(const v8bf*)(Al + ga);
    const size_t gb = (size_t)(n0 + br) * K + k0 + bc;
    bh_r0 = *(const v8bf*)(Bth + gb);
    bh_r1 = *(const v8bf*)(Bth + gb + 8);
    bl_r0 = *(const v8bf*)(Btl + gb);
    bl_r1 = *(const v8bf*)(Btl + gb + 8);
  };
  auto store_tile = [&](int buf) {
    *(v8bf*)&Ash[buf][ar][ac] = ah_r;
    *(v8bf*)&Asl[buf][ar][ac] = al_r;
    *(v8bf*)&Bsh[buf][br][bc] = bh_r0;
    *(v8bf*)&Bsh[buf][br][bc + 8] = bh_r1;
    *(v8bf*)&Bsl[buf][br][bc] = bl_r0;
    *(v8bf*)&Bsl[buf][br][bc + 8] = bl_r1;
  };

  load_tile(0);
  store_tile(0);
  __syncthreads();

  const int niter = K / KB2;
  for (int it = 0; it < niter; ++it) {
    if (it + 1 < niter) load_tile((it + 1) * KB2);  // prefetch to regs
    const int cur = it & 1;

    // Assemble fragments (A and B lane-half K range {h8*16..+15})
    v16bf a_h[2], a_l[2], b_h[2], b_l[2];
#pragma unroll
    for (int mi = 0; mi < 2; ++mi) {
      const int m = wm * 32 + mi * 16 + nl;
      a_h[mi] = cat8(*(const v8bf*)&Ash[cur][m][h8 * 16],
                     *(const v8bf*)&Ash[cur][m][h8 * 16 + 8]);
      a_l[mi] = cat8(*(const v8bf*)&Asl[cur][m][h8 * 16],
                     *(const v8bf*)&Asl[cur][m][h8 * 16 + 8]);
    }
#pragma unroll
    for (int ni = 0; ni < 2; ++ni) {
      const int n = wn * 32 + ni * 16 + nl;
      b_h[ni] = cat8(*(const v8bf*)&Bsh[cur][n][h8 * 16],
                     *(const v8bf*)&Bsh[cur][n][h8 * 16 + 8]);
      b_l[ni] = cat8(*(const v8bf*)&Bsl[cur][n][h8 * 16],
                     *(const v8bf*)&Bsl[cur][n][h8 * 16 + 8]);
    }
#pragma unroll
    for (int mi = 0; mi < 2; ++mi)
#pragma unroll
      for (int ni = 0; ni < 2; ++ni) {
        acc[mi][ni] = wmma_bf16(a_h[mi], b_h[ni], acc[mi][ni]);
        acc[mi][ni] = wmma_bf16(a_h[mi], b_l[ni], acc[mi][ni]);
        acc[mi][ni] = wmma_bf16(a_l[mi], b_h[ni], acc[mi][ni]);
      }

    if (it + 1 < niter) store_tile((it + 1) & 1);  // fill other buffer
    __syncthreads();
  }

#pragma unroll
  for (int mi = 0; mi < 2; ++mi)
#pragma unroll
    for (int ni = 0; ni < 2; ++ni) {
      const int rowb = m0 + wm * 32 + mi * 16 + h8 * 8;
      const int col  = n0 + wn * 32 + ni * 16 + nl;
#pragma unroll
      for (int r = 0; r < 8; ++r)
        Cm[(size_t)(rowb + r) * N + col] = acc[mi][ni][r];
    }
}

// ---------------------------------------------------------------------------
// gate[row, h] = sigmoid(X[row,:] @ W_ilr[:,h] + b_ilr[h]); 32 lanes = 32 heads
// ---------------------------------------------------------------------------
__global__ void ilr_kernel(const float* __restrict__ X,
                           const float* __restrict__ Wilr,
                           const float* __restrict__ bilr,
                           float* __restrict__ gate) {
  const int row = blockIdx.x;
  const int h   = threadIdx.x;  // 0..31
  const float* xr = X + (size_t)row * C_;
  float acc = bilr[h];
  for (int k = 0; k < C_; k += 4) {
    const float4 xv = *(const float4*)(xr + k);  // broadcast across lanes
    acc += xv.x * Wilr[(k + 0) * NH + h];
    acc += xv.y * Wilr[(k + 1) * NH + h];
    acc += xv.z * Wilr[(k + 2) * NH + h];
    acc += xv.w * Wilr[(k + 3) * NH + h];
  }
  gate[(size_t)row * NH + h] = 1.f / (1.f + __expf(-acc));
}

// ---------------------------------------------------------------------------
// TTT scan: one block per (batch, head). W1 (64x64) + b1 stay in LDS over
// the 128-chunk sequential loop. Small matmuls use fp32 WMMA 16x16x4
// (accuracy-critical recurrence, ~5% of total FLOPs). Output is emitted
// directly as bf16 hi/lo planes for the final bf16x3 GEMM.
// ---------------------------------------------------------------------------
__global__ __launch_bounds__(256) void ttt_scan_kernel(
    const float* __restrict__ XC, const float* __restrict__ XB,
    const float* __restrict__ XA, const float* __restrict__ GATE,
    const float* __restrict__ ln_w, const float* __restrict__ ln_b,
    const float* __restrict__ W1_init, const float* __restrict__ b1_init,
    __bf16* __restrict__ OUTh, __bf16* __restrict__ OUTl) {
  __shared__ float W1s[HD][HD + 1];  // persistent fast-weight
  __shared__ float b1s[HD];
  __shared__ float lnw_s[HD], lnb_s[HD];
  __shared__ float xbs[CK][HD + 1];
  __shared__ float xcs[CK][HD + 1];
  __shared__ float xas[CK][HD + 1];
  __shared__ float Z1s[CK][HD + 1];    // xb@W1 + b1
  __shared__ float T1s[CK][HD + 1];    // xc@W1
  __shared__ float grads[CK][HD + 1];
  __shared__ float Gs[CK][HD + 1];     // cumulative grad rows
  __shared__ float Zbars[CK][HD + 1];
  __shared__ float Attns[CK][CK + 1];  // tril(xc@xb^T)
  __shared__ float coeff_s[CK];

  const int tid  = threadIdx.x;
  const int wave = tid >> 5;
  const int lane = tid & 31;
  const int h8 = lane >> 4;
  const int nl = lane & 15;
  const int b  = blockIdx.x / NH;
  const int hH = blockIdx.x % NH;

  const v8f vzero = {0.f, 0.f, 0.f, 0.f, 0.f, 0.f, 0.f, 0.f};

  for (int i = tid; i < HD * HD; i += 256)
    W1s[i / HD][i % HD] = W1_init[(size_t)hH * HD * HD + i];
  if (tid < HD) {
    b1s[tid]   = b1_init[hH * HD + tid];
    lnw_s[tid] = ln_w[hH * HD + tid];
    lnb_s[tid] = ln_b[hH * HD + tid];
  }
  __syncthreads();

  for (int nc = 0; nc < NCHUNK; ++nc) {
    // ---- load chunk tiles (16x64 each) -----------------------------------
    {
      const int t = tid >> 4;
      const int c = (tid & 15) * 4;
      const size_t g = ((size_t)(b * L_ + nc * CK + t)) * C_ + hH * HD + c;
      float4 v;
      v = *(const float4*)(XB + g);
      xbs[t][c] = v.x; xbs[t][c + 1] = v.y; xbs[t][c + 2] = v.z; xbs[t][c + 3] = v.w;
      v = *(const float4*)(XC + g);
      xcs[t][c] = v.x; xcs[t][c + 1] = v.y; xcs[t][c + 2] = v.z; xcs[t][c + 3] = v.w;
      v = *(const float4*)(XA + g);
      xas[t][c] = v.x; xas[t][c + 1] = v.y; xas[t][c + 2] = v.z; xas[t][c + 3] = v.w;
    }
    if (tid < CK) {
      const float sig = GATE[(size_t)(b * L_ + nc * CK + tid) * NH + hH];
      coeff_s[tid] = (1.0f / (float)(tid + 1)) * sig * (1.0f / (float)HD);
    }
    __syncthreads();

    // ---- step 1: Z1 = xb@W1 (+b1), T1 = xc@W1, Attn = tril(xc@xb^T) ------
    if (wave < 4) {
      const int wn = wave;
      v8f acc = vzero;
#pragma unroll
      for (int kb = 0; kb < HD; kb += 4) {
        v2f a, bb;
        a[0]  = xbs[nl][kb + 2 * h8];
        a[1]  = xbs[nl][kb + 2 * h8 + 1];
        bb[0] = W1s[kb + 2 * h8][wn * 16 + nl];
        bb[1] = W1s[kb + 2 * h8 + 1][wn * 16 + nl];
        acc = wmma4(a, bb, acc);
      }
#pragma unroll
      for (int r = 0; r < 8; ++r)
        Z1s[r + h8 * 8][wn * 16 + nl] = acc[r] + b1s[wn * 16 + nl];

      if (wave == 0) {  // Attn: A=xc (16x64), B[d][s]=xb[s][d]
        v8f a2 = vzero;
#pragma unroll
        for (int kb = 0; kb < HD; kb += 4) {
          v2f a, bb;
          a[0]  = xcs[nl][kb + 2 * h8];
          a[1]  = xcs[nl][kb + 2 * h8 + 1];
          bb[0] = xbs[nl][kb + 2 * h8];
          bb[1] = xbs[nl][kb + 2 * h8 + 1];
          a2 = wmma4(a, bb, a2);
        }
#pragma unroll
        for (int r = 0; r < 8; ++r) {
          const int row = r + h8 * 8;
          Attns[row][nl] = (nl <= row) ? a2[r] : 0.f;
        }
      }
    } else {
      const int wn = wave - 4;
      v8f acc = vzero;
#pragma unroll
      for (int kb = 0; kb < HD; kb += 4) {
        v2f a, bb;
        a[0]  = xcs[nl][kb + 2 * h8];
        a[1]  = xcs[nl][kb + 2 * h8 + 1];
        bb[0] = W1s[kb + 2 * h8][wn * 16 + nl];
        bb[1] = W1s[kb + 2 * h8 + 1][wn * 16 + nl];
        acc = wmma4(a, bb, acc);
      }
#pragma unroll
      for (int r = 0; r < 8; ++r)
        T1s[r + h8 * 8][wn * 16 + nl] = acc[r];
    }
    __syncthreads();

    // ---- step 2: grad = ln_fused_l2_bwd(Z1, xa-xb) (one wave per 2 rows) -
#pragma unroll
    for (int rr = 0; rr < 2; ++rr) {
      const int t  = wave * 2 + rr;
      const int d0 = lane * 2;
      const float z0 = Z1s[t][d0], z1 = Z1s[t][d0 + 1];
      const float mu = wave_sum32(z0 + z1) * (1.f / 64.f);
      const float c0 = z0 - mu, c1 = z1 - mu;
      const float var  = wave_sum32(c0 * c0 + c1 * c1) * (1.f / 64.f);
      const float rstd = rsqrtf(var + EPSV);
      const float xh0 = c0 * rstd, xh1 = c1 * rstd;
      const float g0 =
          (lnw_s[d0] * xh0 + lnb_s[d0] - (xas[t][d0] - xbs[t][d0])) * lnw_s[d0];
      const float g1 = (lnw_s[d0 + 1] * xh1 + lnb_s[d0 + 1] -
                        (xas[t][d0 + 1] - xbs[t][d0 + 1])) * lnw_s[d0 + 1];
      const float sg  = wave_sum32(g0 + g1);
      const float sgx = wave_sum32(g0 * xh0 + g1 * xh1);
      const float s = rstd * (1.f / 64.f);
      grads[t][d0]     = (64.f * g0 - sg - xh0 * sgx) * s;
      grads[t][d0 + 1] = (64.f * g1 - sg - xh1 * sgx) * s;
    }
    __syncthreads();

    // ---- step 3: G[t,:] = inclusive cumsum of grad rows ------------------
    if (tid < HD) {
      float run = 0.f;
#pragma unroll
      for (int t = 0; t < CK; ++t) {
        run += grads[t][tid];
        Gs[t][tid] = run;
      }
    }
    __syncthreads();

    // ---- step 4: Zbar = T1 + b1 - co*(Attn@grad + G);  W1 -= coL*xb^T@grad
    const float coL = coeff_s[CK - 1];
    if (wave < 4) {
      const int wn = wave;
      v8f acc = vzero;
#pragma unroll
      for (int kb = 0; kb < CK; kb += 4) {
        v2f a, bb;
        a[0]  = Attns[nl][kb + 2 * h8];
        a[1]  = Attns[nl][kb + 2 * h8 + 1];
        bb[0] = grads[kb + 2 * h8][wn * 16 + nl];
        bb[1] = grads[kb + 2 * h8 + 1][wn * 16 + nl];
        acc = wmma4(a, bb, acc);
      }
#pragma unroll
      for (int r = 0; r < 8; ++r) {
        const int row = r + h8 * 8;
        const int col = wn * 16 + nl;
        Zbars[row][col] =
            T1s[row][col] + b1s[col] - coeff_s[row] * (acc[r] + Gs[row][col]);
      }
    } else {
      const int mt = wave - 4;  // 16-row slab of W1
#pragma unroll
      for (int nt = 0; nt < 4; ++nt) {
        v8f acc = vzero;
#pragma unroll
        for (int kb = 0; kb < CK; kb += 4) {
          v2f a, bb;  // A[m][k] = xb[k][m]
          a[0]  = xbs[kb + 2 * h8][mt * 16 + nl];
          a[1]  = xbs[kb + 2 * h8 + 1][mt * 16 + nl];
          bb[0] = grads[kb + 2 * h8][nt * 16 + nl];
          bb[1] = grads[kb + 2 * h8 + 1][nt * 16 + nl];
          acc = wmma4(a, bb, acc);
        }
#pragma unroll
        for (int r = 0; r < 8; ++r)
          W1s[mt * 16 + r + h8 * 8][nt * 16 + nl] -= coL * acc[r];
      }
    }
    __syncthreads();

    // ---- step 5: b1 update; out = xc + ln_fwd(Zbar) -> bf16 hi/lo --------
    if (tid < HD) b1s[tid] -= coL * Gs[CK - 1][tid];
#pragma unroll
    for (int rr = 0; rr < 2; ++rr) {
      const int t  = wave * 2 + rr;
      const int d0 = lane * 2;
      const float z0 = Zbars[t][d0], z1 = Zbars[t][d0 + 1];
      const float mu = wave_sum32(z0 + z1) * (1.f / 64.f);
      const float c0 = z0 - mu, c1 = z1 - mu;
      const float var  = wave_sum32(c0 * c0 + c1 * c1) * (1.f / 64.f);
      const float rstd = rsqrtf(var + EPSV);
      const float o0 = xcs[t][d0] + lnw_s[d0] * (c0 * rstd) + lnb_s[d0];
      const float o1 =
          xcs[t][d0 + 1] + lnw_s[d0 + 1] * (c1 * rstd) + lnb_s[d0 + 1];
      const size_t g = ((size_t)(b * L_ + nc * CK + t)) * C_ + hH * HD + d0;
      const __bf16 h0 = (__bf16)o0, h1 = (__bf16)o1;
      OUTh[g]     = h0;
      OUTl[g]     = (__bf16)(o0 - (float)h0);
      OUTh[g + 1] = h1;
      OUTl[g + 1] = (__bf16)(o1 - (float)h1);
    }
    __syncthreads();  // protect W1/b1/tile reuse for next chunk
  }
}

// ---------------------------------------------------------------------------
extern "C" void kernel_launch(void* const* d_in, const int* in_sizes, int n_in,
                              void* d_out, int out_size, void* d_ws,
                              size_t ws_size, hipStream_t stream) {
  const float* X    = (const float*)d_in[0];
  const float* Wq   = (const float*)d_in[1];
  const float* Wk   = (const float*)d_in[2];
  const float* Wv   = (const float*)d_in[3];
  const float* Wo   = (const float*)d_in[4];
  const float* Wilr = (const float*)d_in[5];
  const float* bilr = (const float*)d_in[6];
  const float* lnw  = (const float*)d_in[7];
  const float* lnb  = (const float*)d_in[8];
  const float* W1i  = (const float*)d_in[9];
  const float* b1i  = (const float*)d_in[10];

  const size_t nBL = (size_t)B_ * L_;   // 8192
  const size_t nX  = nBL * C_;          // 16.7M elements
  const size_t nW  = (size_t)C_ * C_;   // 4.2M elements

  // fp32 workspace region
  float* XC   = (float*)d_ws;
  float* XB   = XC + nX;
  float* XA   = XB + nX;
  float* GATE = XA + nX;
  // bf16 workspace region (weights stored transposed [N][K])
  __bf16* Xh   = (__bf16*)(GATE + nBL * NH);
  __bf16* Xl   = Xh + nX;
  __bf16* Wqh  = Xl + nX;
  __bf16* Wql  = Wqh + nW;
  __bf16* Wkh  = Wql + nW;
  __bf16* Wkl  = Wkh + nW;
  __bf16* Wvh  = Wkl + nW;
  __bf16* Wvl  = Wvh + nW;
  __bf16* Woh  = Wvl + nW;
  __bf16* Wol  = Woh + nW;
  __bf16* Yh   = Wol + nW;  // scan output (hi)
  __bf16* Yl   = Yh + nX;   // scan output (lo)

  const int M = B_ * L_;
  dim3 gblk(C_ / BN, M / BM);      // 16 x 128
  dim3 gtr(C_ / 32, C_ / 32);      // 64 x 64 transpose tiles

  // fp32 -> bf16 hi/lo splits; weights also transposed (one-time, bandwidth)
  split_bf16<<<(int)(nX / 1024), 256, 0, stream>>>(X, Xh, Xl, (int)nX);
  transpose_split_bf16<<<gtr, 256, 0, stream>>>(Wq, Wqh, Wql, C_, C_);
  transpose_split_bf16<<<gtr, 256, 0, stream>>>(Wk, Wkh, Wkl, C_, C_);
  transpose_split_bf16<<<gtr, 256, 0, stream>>>(Wv, Wvh, Wvl, C_, C_);
  transpose_split_bf16<<<gtr, 256, 0, stream>>>(Wo, Woh, Wol, C_, C_);

  // big GEMMs on the bf16 matrix pipe (error-compensated)
  gemm_wmma_bf16x3<<<gblk, 256, 0, stream>>>(Xh, Xl, Wqh, Wql, XC, M, C_, C_);
  gemm_wmma_bf16x3<<<gblk, 256, 0, stream>>>(Xh, Xl, Wkh, Wkl, XB, M, C_, C_);
  gemm_wmma_bf16x3<<<gblk, 256, 0, stream>>>(Xh, Xl, Wvh, Wvl, XA, M, C_, C_);
  ilr_kernel<<<M, 32, 0, stream>>>(X, Wilr, bilr, GATE);

  ttt_scan_kernel<<<B_ * NH, 256, 0, stream>>>(XC, XB, XA, GATE, lnw, lnb,
                                               W1i, b1i, Yh, Yl);

  gemm_wmma_bf16x3<<<gblk, 256, 0, stream>>>(Yh, Yl, Woh, Wol, (float*)d_out,
                                             M, C_, C_);
}
